// SglKernelAttentionBackend_74491912781952
// MI455X (gfx1250) — matrix operand order repaired and checked by reading, not compiled, not run on hardware
//
#include <hip/hip_runtime.h>

// Flash-attention (causal prefill), B=1, S=2048, H=32, D=128, fp32 in/out.
// S^T = K*Q^T, O^T = V^T*P^T so each lane owns one query column.
// BM=128 (8 waves x 16 queries), BN=64 keys/iteration, bf16 WMMA, fp32 acc.
// Register-prefetch of the next K/V tile hides global latency behind WMMAs.

typedef __bf16 bf16_t;
typedef __attribute__((ext_vector_type(16))) __bf16 v16bf;
typedef __attribute__((ext_vector_type(8)))  float  v8f;
typedef __attribute__((ext_vector_type(4)))  float  v4f;   // clang vector for nontemporal

#define HEADS     32
#define HDIM      128
#define ROWSTRIDE (HEADS * HDIM)   // elements per sequence position
#define BM        128              // query rows per block (8 waves x 16)
#define BN        64               // keys per iteration
#define NWAVES    8
#define NTHREADS  256
#define QT_STRIDE 136              // 128 queries + 8 pad -> 272B rows
#define SK_STRIDE 136              // 128 dims   + 8 pad -> 272B rows
#define VT_STRIDE 72               // 64 keys    + 8 pad -> 144B rows
#define PT_STRIDE 24               // 16 queries + 8 pad -> 48B rows

union FragBF {
  v16bf  v;
  uint4  u[2];
  bf16_t e[16];
};

// sQt is only live before the key loop; overlay it with the loop arrays so
// static LDS stays under 64KB (60,416B total).
struct SmemLoop {
  alignas(16) bf16_t k [BN   * SK_STRIDE];        // K rows: [key][d]
  alignas(16) bf16_t vt[HDIM * VT_STRIDE];        // V^T: [d][key]
  alignas(16) bf16_t pt[NWAVES * BN * PT_STRIDE]; // per-wave P^T: [key][q]
};
union Smem {
  alignas(16) bf16_t qt[HDIM * QT_STRIDE];        // Q^T: [d][qlocal]
  SmemLoop loop;
};

__device__ __forceinline__ bf16_t f2bf(float f) { return (bf16_t)f; }

// Packed pair: two floats -> one dword of two bf16.
__device__ __forceinline__ unsigned f2bf2(float a, float b) {
  union { bf16_t e[2]; unsigned u; } x;
  x.e[0] = (bf16_t)a;
  x.e[1] = (bf16_t)b;
  return x.u;
}

__global__ __launch_bounds__(NTHREADS, 1)
void fa2_causal_bf16_wmma(const float* __restrict__ Q,
                          const float* __restrict__ K,
                          const float* __restrict__ V,
                          float* __restrict__ Out) {
  __shared__ Smem sm;

  const int tid  = threadIdx.x;
  const int wave = tid >> 5;
  const int lane = tid & 31;
  const int r16  = lane & 15;   // row within 16-row A tile / C-tile column
  const int hi   = lane >> 4;

  const int h     = blockIdx.y;
  const int qbase = blockIdx.x * BM;
  const int qwave = qbase + wave * 16;   // first query row of this wave
  const int qlane = qwave + r16;         // this lane's query (C-tile column)

  // softmax scale 1/sqrt(128) and log2(e) folded into Q (exp -> exp2 domain)
  const float qscale = 0.08838834764831845f * 1.4426950408889634f;

  // ---- Stage Q^T once (scaled bf16): thread owns 2 adjacent queries x 32
  // dims so transposed stores are packed b32 pairs. ----
  {
    const int qp2    = (tid & 63) * 2;  // query pair base (0..126)
    const int dchk   = tid >> 6;        // 0..3 -> 32 dims
    const float* q0 = Q + (size_t)(qbase + qp2) * ROWSTRIDE + h * HDIM + dchk * 32;
    const float* q1 = q0 + ROWSTRIDE;
    alignas(16) float a[32];
    alignas(16) float b[32];
#pragma unroll
    for (int i = 0; i < 8; ++i) *(float4*)(a + 4 * i) = *(const float4*)(q0 + 4 * i);
#pragma unroll
    for (int i = 0; i < 8; ++i) *(float4*)(b + 4 * i) = *(const float4*)(q1 + 4 * i);
#pragma unroll
    for (int i = 0; i < 32; ++i)
      *(unsigned*)&sm.qt[(dchk * 32 + i) * QT_STRIDE + qp2] =
          f2bf2(a[i] * qscale, b[i] * qscale);
  }
  __syncthreads();

  // ---- Loop-invariant B fragments of Q^T (lane = query column r16, 16 dims
  // per lane in table k order) ----
  FragBF qtf[4];
#pragma unroll
  for (int kc = 0; kc < 4; ++kc) {
#pragma unroll
    for (int s = 0; s < 16; ++s)
      qtf[kc].e[s] = sm.qt[(kc * 32 + (s < 8 ? hi * 8 + s : 16 + hi * 8 + (s - 8))) * QT_STRIDE + wave * 16 + r16];
  }
  __syncthreads();   // qt region is overlaid by the loop arrays below

  // ---- O^T accumulators: tile t covers dims t*16..t*16+15; col = query ----
  v8f o[8];
#pragma unroll
  for (int t = 0; t < 8; ++t) {
    v8f z = {0.f, 0.f, 0.f, 0.f, 0.f, 0.f, 0.f, 0.f};
    o[t] = z;
  }
  float m = -1e30f, l = 0.f;   // per-lane (per-query) softmax stats

  bf16_t* pw = &sm.loop.pt[wave * BN * PT_STRIDE];
  const int kend = qbase + BM;

  // ---- Staging geometry: thread owns 2 adjacent keys x 16 dims ----
  const int kp2  = (tid & 31) * 2;  // key pair base (0..62)
  const int dchk = tid >> 5;        // 0..7 -> 16 dims
  const float* kbase = K + (size_t)kp2 * ROWSTRIDE + h * HDIM + dchk * 16;
  const float* vbase = V + (size_t)kp2 * ROWSTRIDE + h * HDIM + dchk * 16;

  // Prefetch registers for the current K/V tile.
  alignas(16) float pka[16];
  alignas(16) float pkb[16];
  alignas(16) float pva[16];
  alignas(16) float pvb[16];
  {
    const float* k0 = kbase;              // j0 = 0
    const float* v0 = vbase;
#pragma unroll
    for (int i = 0; i < 4; ++i) *(float4*)(pka + 4 * i) = *(const float4*)(k0 + 4 * i);
#pragma unroll
    for (int i = 0; i < 4; ++i) *(float4*)(pkb + 4 * i) = *(const float4*)(k0 + ROWSTRIDE + 4 * i);
#pragma unroll
    for (int i = 0; i < 4; ++i) *(float4*)(pva + 4 * i) = *(const float4*)(v0 + 4 * i);
#pragma unroll
    for (int i = 0; i < 4; ++i) *(float4*)(pvb + 4 * i) = *(const float4*)(v0 + ROWSTRIDE + 4 * i);
  }

  for (int j0 = 0; j0 < kend; j0 += BN) {
    // ---- Convert prefetched tile and store to LDS ----
    {
      alignas(16) bf16_t kt[32];
#pragma unroll
      for (int i = 0; i < 16; ++i) kt[i]      = f2bf(pka[i]);
#pragma unroll
      for (int i = 0; i < 16; ++i) kt[16 + i] = f2bf(pkb[i]);
#pragma unroll
      for (int i = 0; i < 2; ++i)
        *(uint4*)(&sm.loop.k[kp2 * SK_STRIDE + dchk * 16 + i * 8]) =
            *(const uint4*)(&kt[i * 8]);
#pragma unroll
      for (int i = 0; i < 2; ++i)
        *(uint4*)(&sm.loop.k[(kp2 + 1) * SK_STRIDE + dchk * 16 + i * 8]) =
            *(const uint4*)(&kt[16 + i * 8]);
#pragma unroll
      for (int i = 0; i < 16; ++i)
        *(unsigned*)&sm.loop.vt[(dchk * 16 + i) * VT_STRIDE + kp2] = f2bf2(pva[i], pvb[i]);
    }
    __syncthreads();

    // ---- Prefetch next tile (clamped on last iteration); issues during
    //      the compute phase below, waited at next iteration's convert ----
    {
      int jn = j0 + BN;
      if (jn >= kend) jn = kend - BN;
      const float* k0 = kbase + (size_t)jn * ROWSTRIDE;
      const float* v0 = vbase + (size_t)jn * ROWSTRIDE;
#pragma unroll
      for (int i = 0; i < 4; ++i) *(float4*)(pka + 4 * i) = *(const float4*)(k0 + 4 * i);
#pragma unroll
      for (int i = 0; i < 4; ++i) *(float4*)(pkb + 4 * i) = *(const float4*)(k0 + ROWSTRIDE + 4 * i);
#pragma unroll
      for (int i = 0; i < 4; ++i) *(float4*)(pva + 4 * i) = *(const float4*)(v0 + 4 * i);
#pragma unroll
      for (int i = 0; i < 4; ++i) *(float4*)(pvb + 4 * i) = *(const float4*)(v0 + ROWSTRIDE + 4 * i);
    }

    // Skip fully-masked key blocks for this wave (uniform per wave at runtime).
    if (j0 <= qwave + 15) {
      // ---- S^T = K*Q^T : four 16(key)x16(query) tiles, chained over D ----
      float s[4][8];
#pragma unroll
      for (int t = 0; t < 4; ++t) {
        v8f c = {0.f, 0.f, 0.f, 0.f, 0.f, 0.f, 0.f, 0.f};
#pragma unroll
        for (int kc = 0; kc < 4; ++kc) {
          FragBF a;  // A layout: lane row = key (r16), two 8-dim chunks
          const bf16_t* src = &sm.loop.k[(t * 16 + r16) * SK_STRIDE + kc * 32 + hi * 8];
          a.u[0] = *(const uint4*)(src);
          a.u[1] = *(const uint4*)(src + 16);
          c = __builtin_amdgcn_wmma_f32_16x16x32_bf16(
                false, a.v, false, qtf[kc].v, (short)0, c, false, false);
        }
#pragma unroll
        for (int r = 0; r < 8; ++r) s[t][r] = c[r];
      }

      // ---- Causal mask: key > query -> -inf (only near the diagonal) ----
      if (j0 + BN - 1 > qwave) {
#pragma unroll
        for (int t = 0; t < 4; ++t)
#pragma unroll
          for (int r = 0; r < 8; ++r)
            if (j0 + t * 16 + hi * 8 + r > qlane) s[t][r] = -1e30f;
      }

      // ---- Per-lane online softmax; balanced-tree reductions over 32 ----
      float f[32];
#pragma unroll
      for (int t = 0; t < 4; ++t)
#pragma unroll
        for (int r = 0; r < 8; ++r) f[t * 8 + r] = s[t][r];
#pragma unroll
      for (int st = 16; st >= 1; st >>= 1)
#pragma unroll
        for (int i = 0; i < 32; ++i)
          if (i < st) f[i] = fmaxf(f[i], f[i + st]);
      const float mx    = fmaxf(f[0], __shfl_xor(f[0], 16, 32));
      const float mnew  = fmaxf(m, mx);
      const float alpha = __builtin_amdgcn_exp2f(m - mnew);
      float p[32];
#pragma unroll
      for (int t = 0; t < 4; ++t)
#pragma unroll
        for (int r = 0; r < 8; ++r)
          p[t * 8 + r] = __builtin_amdgcn_exp2f(s[t][r] - mnew);
      float g[32];
#pragma unroll
      for (int i = 0; i < 32; ++i) g[i] = p[i];
#pragma unroll
      for (int st = 16; st >= 1; st >>= 1)
#pragma unroll
        for (int i = 0; i < 32; ++i)
          if (i < st) g[i] += g[i + st];
      const float rs = g[0] + __shfl_xor(g[0], 16, 32);
      l = l * alpha + rs;
      m = mnew;
#pragma unroll
      for (int t = 0; t < 8; ++t) o[t] *= alpha;

      // ---- Stage P^T as [key][query] (wave-private slab, in-order LDS) ----
#pragma unroll
      for (int t = 0; t < 4; ++t)
#pragma unroll
        for (int r = 0; r < 8; ++r)
          pw[(t * 16 + hi * 8 + r) * PT_STRIDE + r16] = f2bf(p[t * 8 + r]);
      FragBF pf[2];  // two B fragments (keys 0..31, 32..63), each reused 8x
#pragma unroll
      for (int kk = 0; kk < 2; ++kk) {
#pragma unroll
        for (int s = 0; s < 16; ++s)
          pf[kk].e[s] = pw[(kk * 32 + (s < 8 ? hi * 8 + s : 16 + hi * 8 + (s - 8))) * PT_STRIDE + r16];
      }

      // ---- O^T += V^T * P^T : 16 WMMAs over D=128, K=64 ----
#pragma unroll
      for (int t = 0; t < 8; ++t) {
        v8f c = o[t];
#pragma unroll
        for (int kk = 0; kk < 2; ++kk) {
          FragBF a;  // A layout: lane row = dim (t*16+r16), two 8-key chunks
          const bf16_t* vsrc =
              &sm.loop.vt[(t * 16 + r16) * VT_STRIDE + kk * 32 + hi * 8];
          a.u[0] = *(const uint4*)(vsrc);
          a.u[1] = *(const uint4*)(vsrc + 16);
          c = __builtin_amdgcn_wmma_f32_16x16x32_bf16(
                false, a.v, false, pf[kk].v, (short)0, c, false, false);
        }
        o[t] = c;
      }
    }
    __syncthreads();   // protect sK/sVt before next iteration's staging
  }

  // ---- Normalize, write out [B, S, H*D]; 8 contiguous dims per lane-tile
  //      -> vectorized nontemporal b128 stores ----
  const float inv = __builtin_amdgcn_rcpf(l);
  float* orow = Out + (size_t)qlane * ROWSTRIDE + h * HDIM;
#pragma unroll
  for (int t = 0; t < 8; ++t) {
    union { float f[8]; v4f q[2]; } t8;
#pragma unroll
    for (int r = 0; r < 8; ++r) t8.f[r] = o[t][r] * inv;
    __builtin_nontemporal_store(t8.q[0], (v4f*)(orow + t * 16 + hi * 8));
    __builtin_nontemporal_store(t8.q[1], (v4f*)(orow + t * 16 + hi * 8 + 4));
  }
}

extern "C" void kernel_launch(void* const* d_in, const int* in_sizes, int n_in,
                              void* d_out, int out_size, void* d_ws, size_t ws_size,
                              hipStream_t stream) {
  (void)n_in; (void)out_size; (void)d_ws; (void)ws_size;
  const float* q = (const float*)d_in[0];
  const float* k = (const float*)d_in[1];
  const float* v = (const float*)d_in[2];
  float* out = (float*)d_out;
  const int S = in_sizes[0] / ROWSTRIDE;   // 2048 for the reference shapes
  dim3 grid((unsigned)(S / BM), HEADS, 1);
  fa2_causal_bf16_wmma<<<grid, NTHREADS, 0, stream>>>(q, k, v, out);
}
